// GCLIQuantizedLinear_64330020159902
// MI455X (gfx1250) — matrix operand order repaired and checked by reading, not compiled, not run on hardware
//
#include <hip/hip_runtime.h>
#include <hip/hip_bf16.h>

// Problem dimensions (fixed by the reference)
#define M_DIM 8192   // B*S rows of x / out
#define N_DIM 4096   // reduction dimension (columns of x and W)
#define K_DIM 4096   // output features (rows of W)
#define N_GROUPS 64  // N_DIM / 64 quant groups per W row

#define CHUNK_K 64                // K-depth staged per buffer
#define NCHUNK (N_DIM / CHUNK_K)  // 64
#define XTILE_BYTES 16384         // 128 rows x 64 K x 2B, fragment-linear
#define BUF_BYTES   32768         // X tile + W tile

typedef __attribute__((ext_vector_type(16))) __bf16         bf16x16;
typedef __attribute__((ext_vector_type(8)))  __bf16         bf16x8;
typedef __attribute__((ext_vector_type(8)))  float          f32x8;
typedef __attribute__((ext_vector_type(4)))  unsigned int   u32x4;
typedef __attribute__((ext_vector_type(8)))  unsigned short u16x8;

__device__ __forceinline__ unsigned short f32_to_bf16_rne(float x) {
  unsigned int u = __builtin_bit_cast(unsigned int, x);
  u += 0x7FFFu + ((u >> 16) & 1u);   // round-to-nearest-even
  return (unsigned short)(u >> 16);
}

// ---------------------------------------------------------------------------
// Kernel 1: dequantize W_q -> bf16, folding scales, zeros, mu2(row), mu1(col).
// ---------------------------------------------------------------------------
__global__ void gcli_dequant_w(const int* __restrict__ Wq,
                               const float* __restrict__ scales,
                               const float* __restrict__ zeros,
                               const float* __restrict__ mu1,
                               const float* __restrict__ mu2,
                               unsigned short* __restrict__ Wbf) {
  const int idx = blockIdx.x * 256 + threadIdx.x;   // [0, K_DIM * N_DIM/8)
  const int k   = idx >> 9;                         // N_DIM/8 = 512 per row
  const int n0  = (idx & 511) << 3;
  const int g   = n0 >> 6;

  const float s = scales[k * N_GROUPS + g];
  const float z = zeros[k * N_GROUPS + g];
  const float f = s * mu2[k];

  const int*   q  = Wq  + (long)k * N_DIM + n0;
  const float* m1 = mu1 + n0;

  u16x8 o;
#pragma unroll
  for (int i = 0; i < 8; ++i) {
    float w = ((float)q[i] - z) * f * m1[i];
    o[i] = f32_to_bf16_rne(w);
  }
  *(u16x8*)(Wbf + (long)k * N_DIM + n0) = o;
}

// ---------------------------------------------------------------------------
// Kernel 2: convert x f32 -> bf16 (RNE).
// ---------------------------------------------------------------------------
__global__ void gcli_convert_x(const float* __restrict__ X,
                               unsigned short* __restrict__ Xbf) {
  const long idx = ((long)blockIdx.x * 256 + threadIdx.x) * 8;
  const float* p = X + idx;
  u16x8 o;
#pragma unroll
  for (int i = 0; i < 8; ++i) o[i] = f32_to_bf16_rne(p[i]);
  *(u16x8*)(Xbf + idx) = o;
}

// ---------------------------------------------------------------------------
// Fragment read from LDS (fragment-linear layout): lane's 32 contiguous bytes.
// ---------------------------------------------------------------------------
__device__ __forceinline__ bf16x16 gcli_lds_frag(const unsigned char* p) {
  bf16x8 lo = __builtin_bit_cast(bf16x8, *(const u32x4*)p);
  bf16x8 hi = __builtin_bit_cast(bf16x8, *(const u32x4*)(p + 16));
  bf16x16 f;
#pragma unroll
  for (int i = 0; i < 8; ++i) { f[i] = lo[i]; f[i + 8] = hi[i]; }
  return f;
}

// ---------------------------------------------------------------------------
// Kernel 3: bf16 WMMA GEMM, async-to-LDS double-buffered.
//
// Block: 256 threads = 8 wave32, block tile 128(m) x 128(kout).
// Wave tile: 64 x 32 = 4x2 accumulators of 16x16 f32.
//
// LDS holds tiles in FRAGMENT-LINEAR form: each 16(row) x 32(K) operand is a
// 1 KB block where lane l's 16 bf16 values live at [lane*32 .. lane*32+31].
// The global->LDS scatter is done by global_load_async_to_lds_b128 with a
// per-lane permuted LDS destination:
//   A piece (c8 = (col/8)%4):  sub-lane group hi = c8&1, half = c8>>1
//   B piece:                   hi = c8&1, half = c8>>1
//   lds_off(frag) = (hi*16 + l16)*32 + half*16
// This reproduces exactly the V_WMMA_16X16X32_BF16 A/B register layouts and
// makes the ds_load_b128 reads bank-conflict-free with no padding.
// ---------------------------------------------------------------------------
__global__ void __launch_bounds__(256)
gcli_wmma_gemm(const __bf16* __restrict__ Xbf,
               const __bf16* __restrict__ Wbf,
               const float* __restrict__ bias,
               float* __restrict__ out) {
  __shared__ unsigned char smem[2 * BUF_BYTES];   // 64 KB, double buffered

  const int tid   = threadIdx.x;
  const int lane  = tid & 31;
  const int l16   = lane & 15;
  const int hiC   = lane >> 4;     // C/D row sub-offset (0 or 8)
  const int wave  = tid >> 5;      // 0..7
  const int waveM = wave & 1;      // 2 waves along m
  const int waveN = wave >> 1;     // 4 waves along kout

  const int mBase = blockIdx.y * 128;
  const int kBase = blockIdx.x * 128;

  // LDS base as a raw 32-bit offset (flat LDS aperture: low 32 bits = offset)
  const unsigned sbase = (unsigned)(size_t)smem;

  // ---- per-thread async-copy piece descriptors (4 X pieces + 4 W pieces) ----
  unsigned      ldsX[4], ldsW[4];
  const __bf16* gX[4];
  const __bf16* gW[4];
#pragma unroll
  for (int j = 0; j < 4; ++j) {
    const int c   = j * 256 + tid;   // 16B piece index in [0,1024)
    const int row = c >> 3;          // tile row 0..127
    const int p   = c & 7;           // 16B piece within the 64-K row
    const int kb  = p >> 2;          // 32-K sub-block 0..1
    const int c8  = p & 3;           // 8-element piece within 32-K block
    const int rb  = row >> 4;        // 16-row fragment block 0..7
    const int rl  = row & 15;

    const unsigned fragBase = (unsigned)(((rb * 2 + kb) << 10));
    const int hiX = c8 & 1, hfX = c8 >> 1;             // A-operand mapping
    const int hiW = c8 & 1, hfW = c8 >> 1;             // B-operand mapping
    ldsX[j] = fragBase + (unsigned)((hiX * 16 + rl) * 32 + hfX * 16);
    ldsW[j] = XTILE_BYTES + fragBase + (unsigned)((hiW * 16 + rl) * 32 + hfW * 16);
    gX[j]   = Xbf + (long)(mBase + row) * N_DIM + p * 8;
    gW[j]   = Wbf + (long)(kBase + row) * N_DIM + p * 8;
  }

  // ---- pre-issue chunks 0 and 1 ----
#pragma unroll
  for (int pre = 0; pre < 2; ++pre) {
    const unsigned sb = sbase + (unsigned)(pre * BUF_BYTES);
    const long nOff = (long)pre * CHUNK_K;
#pragma unroll
    for (int j = 0; j < 4; ++j) {
      asm volatile("global_load_async_to_lds_b128 %0, %1, off"
                   :: "v"(sb + ldsX[j]), "v"(gX[j] + nOff) : "memory");
      asm volatile("global_load_async_to_lds_b128 %0, %1, off"
                   :: "v"(sb + ldsW[j]), "v"(gW[j] + nOff) : "memory");
    }
  }

  f32x8 acc[4][2] = {};

  for (int chunk = 0; chunk < NCHUNK; ++chunk) {
    // Async loads complete in issue order: <=8 outstanding means the current
    // buffer's 8 transfers are resident.
    if (chunk + 1 < NCHUNK)
      asm volatile("s_wait_asynccnt 0x8" ::: "memory");
    else
      asm volatile("s_wait_asynccnt 0x0" ::: "memory");
    __syncthreads();

    const unsigned char* bufp = smem + (chunk & 1) * BUF_BYTES;

#pragma unroll
    for (int kb = 0; kb < 2; ++kb) {
      bf16x16 a[4], b[2];
#pragma unroll
      for (int mi = 0; mi < 4; ++mi)
        a[mi] = gcli_lds_frag(bufp + (((waveM * 4 + mi) * 2 + kb) << 10) + lane * 32);
#pragma unroll
      for (int ni = 0; ni < 2; ++ni)
        b[ni] = gcli_lds_frag(bufp + XTILE_BYTES +
                              (((waveN * 2 + ni) * 2 + kb) << 10) + lane * 32);

#pragma unroll
      for (int mi = 0; mi < 4; ++mi)
#pragma unroll
        for (int ni = 0; ni < 2; ++ni)
          acc[mi][ni] = __builtin_amdgcn_wmma_f32_16x16x32_bf16(
              false, a[mi], false, b[ni], (short)0, acc[mi][ni], false, false);
    }

    __syncthreads();   // everyone done reading buf[chunk&1] before overwrite

    if (chunk + 2 < NCHUNK) {
      const unsigned sb = sbase + (unsigned)((chunk & 1) * BUF_BYTES);
      const long nOff = (long)(chunk + 2) * CHUNK_K;
#pragma unroll
      for (int j = 0; j < 4; ++j) {
        asm volatile("global_load_async_to_lds_b128 %0, %1, off"
                     :: "v"(sb + ldsX[j]), "v"(gX[j] + nOff) : "memory");
        asm volatile("global_load_async_to_lds_b128 %0, %1, off"
                     :: "v"(sb + ldsW[j]), "v"(gW[j] + nOff) : "memory");
      }
    }
  }

  // ---- Epilogue: C/D layout -> VGPR r = row (r + 8*hiC), lane l16 = col ----
#pragma unroll
  for (int ni = 0; ni < 2; ++ni) {
    const int col = kBase + waveN * 32 + ni * 16 + l16;
    const float bv = bias[col];
#pragma unroll
    for (int mi = 0; mi < 4; ++mi) {
#pragma unroll
      for (int r = 0; r < 8; ++r) {
        const int row = mBase + waveM * 64 + mi * 16 + hiC * 8 + r;
        out[(long)row * K_DIM + col] = acc[mi][ni][r] + bv;
      }
    }
  }
}

// ---------------------------------------------------------------------------
// Launch
// ---------------------------------------------------------------------------
extern "C" void kernel_launch(void* const* d_in, const int* in_sizes, int n_in,
                              void* d_out, int out_size, void* d_ws, size_t ws_size,
                              hipStream_t stream) {
  const float* x      = (const float*)d_in[0];
  const float* scales = (const float*)d_in[1];
  const float* zeros  = (const float*)d_in[2];
  const float* mu1    = (const float*)d_in[3];
  const float* mu2    = (const float*)d_in[4];
  const float* bias   = (const float*)d_in[5];
  const int*   Wq     = (const int*)d_in[6];
  float*       out    = (float*)d_out;

  // Workspace layout: [ Wbf16 : 32 MiB ][ Xbf16 : 64 MiB ]
  unsigned short* Wbf = (unsigned short*)d_ws;
  unsigned short* Xbf = Wbf + (size_t)K_DIM * N_DIM;

  gcli_dequant_w<<<(K_DIM * (N_DIM / 8)) / 256, 256, 0, stream>>>(
      Wq, scales, zeros, mu1, mu2, Wbf);

  gcli_convert_x<<<(int)(((long)M_DIM * N_DIM / 8) / 256), 256, 0, stream>>>(
      x, Xbf);

  dim3 grid(K_DIM / 128, M_DIM / 128);
  gcli_wmma_gemm<<<grid, 256, 0, stream>>>(
      (const __bf16*)Xbf, (const __bf16*)Wbf, bias, out);
}
